// GRU_Precoder_53850299957345
// MI455X (gfx1250) — matrix operand order repaired and checked by reading, not compiled, not run on hardware
//
#include <hip/hip_runtime.h>

typedef __attribute__((ext_vector_type(16))) __bf16 v16bf;
typedef __attribute__((ext_vector_type(8)))  float  v8f;
typedef __attribute__((ext_vector_type(4)))  unsigned int u32x4;
typedef __attribute__((ext_vector_type(4)))  float  f32x4;
typedef unsigned short u16;

#define B_    1024
#define OBS_  128
#define ZDIM_ 64
#define DIN_  192
#define HID_  512
#define CTX_  64
#define ACT_  32
#define MLPH_ 256
#define TH3_  1536

// workspace layout (bytes, all 256-aligned)
#define OFF_WHSWZ 0u
#define OFF_WOSWZ (OFF_WHSWZ + 512u*1536u*2u)   // 1572864
#define OFF_X0    (OFF_WOSWZ + 512u*32u*2u)     // 1605632
#define OFF_H1    (OFF_X0 + 1024u*192u*4u)      // 2392064
#define OFF_H2    (OFF_H1 + 1024u*256u*4u)      // 3440640
#define OFF_H0    (OFF_H2 + 1024u*256u*4u)      // 4489216
#define OFF_XIT   (OFF_H0 + 1024u*512u*4u)      // 6586368
// total ~12.9 MB

__device__ __forceinline__ u16 f2bf(float f) {
  unsigned int u = __float_as_uint(f);
  u += 0x7FFFu + ((u >> 16) & 1u);   // round-to-nearest-even
  return (u16)(u >> 16);
}
__device__ __forceinline__ float sigm(float x) { return 1.0f / (1.0f + __expf(-x)); }

union FragAB { v16bf v; u32x4 q[2]; u16 h[16]; };
union F8     { f32x4 q[2]; float f[8]; };

// ---- x0 = concat(s_t[:,0,:], z_t) ----------------------------------------
__global__ void k_x0(const float* __restrict__ s_t, const float* __restrict__ z_t,
                     float* __restrict__ X0) {
  int b = blockIdx.x, t = threadIdx.x;                 // block = 192 threads
  X0[b*DIN_ + t] = (t < OBS_) ? s_t[(size_t)b*64*OBS_ + t]
                              : z_t[b*ZDIM_ + (t - OBS_)];
}

// ---- pre-swizzle Wh (f32 [512,1536]) -> bf16 B-fragments ------------------
// fragment (kt,nt): lane l holds 16 contiguous bf16: K = kt*32 + 16*(l>>4) + e,
// N = nt*16 + (l&15). One 32B store per lane; one-shot, runs once per call.
__global__ __launch_bounds__(256) void k_swz_wh(const float* __restrict__ Wh,
                                                u16* __restrict__ out) {
  int idx = blockIdx.x*256 + threadIdx.x;              // 16*96*32 = 49152
  int lane = idx & 31, frag = idx >> 5;
  int nt = frag % 96, kt = frag / 96;
  int n  = nt*16 + (lane & 15);
  int kb = kt*32 + ((lane >> 4) << 4);
  FragAB f;
#pragma unroll
  for (int e = 0; e < 16; ++e) f.h[e] = f2bf(Wh[(size_t)(kb + e)*TH3_ + n]);
  u32x4* dst = (u32x4*)(out + (size_t)idx*16);
  dst[0] = f.q[0]; dst[1] = f.q[1];
}

// ---- pre-swizzle Wo (f32 [512,32]) -> bf16 B-fragments --------------------
__global__ __launch_bounds__(256) void k_swz_wo(const float* __restrict__ Wo,
                                                u16* __restrict__ out) {
  int idx = blockIdx.x*256 + threadIdx.x;              // 16*2*32 = 1024
  int lane = idx & 31, frag = idx >> 5;
  int nt = frag & 1, kt = frag >> 1;
  int n  = nt*16 + (lane & 15);
  int kb = kt*32 + ((lane >> 4) << 4);
  FragAB f;
#pragma unroll
  for (int e = 0; e < 16; ++e) f.h[e] = f2bf(Wo[(size_t)(kb + e)*ACT_ + n]);
  u32x4* dst = (u32x4*)(out + (size_t)idx*16);
  dst[0] = f.q[0]; dst[1] = f.q[1];
}

// ---- Dense + LayerNorm + ReLU (one block per batch row) -------------------
__global__ __launch_bounds__(256) void k_mlp_ln(const float* __restrict__ X, int Kdim,
                                                const float* __restrict__ W,
                                                const float* __restrict__ bv,
                                                const float* __restrict__ g,
                                                const float* __restrict__ be,
                                                float* __restrict__ Y) {
  __shared__ float xrow[256];
  __shared__ float s1[256], s2[256];
  int row = blockIdx.x, tid = threadIdx.x;
  if (tid < Kdim) xrow[tid] = X[row*Kdim + tid];
  __syncthreads();
  float acc = bv[tid];
  for (int k = 0; k < Kdim; ++k) acc += xrow[k] * W[k*MLPH_ + tid];
  s1[tid] = acc; s2[tid] = acc*acc;
  __syncthreads();
  for (int off = 128; off > 0; off >>= 1) {
    if (tid < off) { s1[tid] += s1[tid+off]; s2[tid] += s2[tid+off]; }
    __syncthreads();
  }
  float m = s1[0] * (1.0f/256.0f);
  float v = s2[0] * (1.0f/256.0f) - m*m;
  float y = (acc - m) * rsqrtf(v + 1e-6f) * g[tid] + be[tid];
  Y[row*MLPH_ + tid] = y > 0.0f ? y : 0.0f;
}

// ---- h0 = h2 @ W3 + b3 ----------------------------------------------------
__global__ __launch_bounds__(256) void k_h0(const float* __restrict__ H2,
                                            const float* __restrict__ W3,
                                            const float* __restrict__ b3,
                                            float* __restrict__ H0) {
  __shared__ float xrow[256];
  int row = blockIdx.x, tid = threadIdx.x;
  xrow[tid] = H2[row*MLPH_ + tid];
  __syncthreads();
  for (int cc = 0; cc < 2; ++cc) {
    int col = tid + cc*256;
    float acc = b3[col];
    for (int k = 0; k < MLPH_; ++k) acc += xrow[k] * W3[k*HID_ + col];
    H0[row*HID_ + col] = acc;
  }
}

// ---- xi_t[j][b] = dot(x0[b,:], Wi[:,j])  (transposed for coalesced loads) -
__global__ __launch_bounds__(256) void k_xi(const float* __restrict__ X0,
                                            const float* __restrict__ Wi,
                                            float* __restrict__ XiT) {
  __shared__ float wcol[DIN_];
  int j = blockIdx.x, tid = threadIdx.x;
  if (tid < DIN_) wcol[tid] = Wi[(size_t)tid*TH3_ + j];
  __syncthreads();
  for (int c = 0; c < 4; ++c) {
    int b = c*256 + tid;
    float acc = 0.0f;
    const float* xr = X0 + (size_t)b*DIN_;
    for (int k = 0; k < DIN_; ++k) acc += xr[k] * wcol[k];
    XiT[(size_t)j*B_ + b] = acc;
  }
}

// ---- Fused 64-step GRU + output head. 32 blocks x 256 threads (8 waves). --
// Block owns 32 batch rows. fp32 carry lives in VGPRs; bf16 h double-buffered
// in LDS for WMMA A-fragments. Wave w owns hidden N-tiles {w, w+8, w+16, w+24}.
__global__ __launch_bounds__(256) void k_gru(const float* __restrict__ H0,
                                             const float* __restrict__ XiT,
                                             const float* __restrict__ bh,
                                             const u16* __restrict__ wh_swz,
                                             const u16* __restrict__ wo_swz,
                                             const float* __restrict__ bo,
                                             float* __restrict__ out) {
  __shared__ u16 hbuf[2][32][520];                 // 520-u16 rows: 1040B = 65*16B
  const int tid  = threadIdx.x;
  const int w    = tid >> 5;
  const int lane = tid & 31;
  const int half = lane >> 4;
  const int lcol = lane & 15;
  const int bBase = blockIdx.x * 32;
  const v8f vzero = {};

  float hreg[2][4][8];                             // fp32 carry: [mtile][ntile][elem]
  float bhr[4], bhz[4], bhn[4];

#pragma unroll
  for (int ntl = 0; ntl < 4; ++ntl) {
    const int nt = w + ntl*8;
    const int colh = nt*16 + lcol;
    bhr[ntl] = bh[colh];
    bhz[ntl] = bh[HID_ + colh];
    bhn[ntl] = bh[2*HID_ + colh];
#pragma unroll
    for (int mt = 0; mt < 2; ++mt)
#pragma unroll
      for (int r = 0; r < 8; ++r) {
        const int row = bBase + mt*16 + 8*half + r;
        const float h = H0[(size_t)row*HID_ + colh];
        hreg[mt][ntl][r] = h;
        hbuf[0][mt*16 + 8*half + r][colh] = f2bf(h);
      }
  }
  const float bo_l = bo[(w & 1)*16 + lcol];
  __syncthreads();

  int cur = 0;
  for (int t = 0; t < CTX_; ++t) {
    const int nxt = cur ^ 1;

    // ---- phase 1: gh = h @ Wh (3 gates), GRU elementwise update ----
#pragma unroll
    for (int ntl = 0; ntl < 4; ++ntl) {
      const int nt = w + ntl*8;
      const int colh = nt*16 + lcol;
      v8f accr[2], accz[2], accn[2];
#pragma unroll
      for (int mt = 0; mt < 2; ++mt) { accr[mt] = vzero; accz[mt] = vzero; accn[mt] = vzero; }

      for (int kt = 0; kt < 16; ++kt) {
        const int kb = kt*32 + 16*half;
        FragAB a0, a1, brf, bzf, bnf;
        const u16* ap0 = &hbuf[cur][lcol][kb];
        const u16* ap1 = &hbuf[cur][16 + lcol][kb];
        a0.q[0] = *(const u32x4*)(ap0); a0.q[1] = *(const u32x4*)(ap0 + 8);
        a1.q[0] = *(const u32x4*)(ap1); a1.q[1] = *(const u32x4*)(ap1 + 8);
        const u16* bp = wh_swz + ((size_t)(kt*96 + nt)*32 + lane)*16;
        const u16* zp = bp + 32*512;               // +32 N-tiles (z-gate columns)
        const u16* np = bp + 64*512;               // +64 N-tiles (n-gate columns)
        brf.q[0] = *(const u32x4*)(bp); brf.q[1] = *(const u32x4*)(bp + 8);
        bzf.q[0] = *(const u32x4*)(zp); bzf.q[1] = *(const u32x4*)(zp + 8);
        bnf.q[0] = *(const u32x4*)(np); bnf.q[1] = *(const u32x4*)(np + 8);

        accr[0] = __builtin_amdgcn_wmma_f32_16x16x32_bf16(false, a0.v, false, brf.v, (short)0, accr[0], false, false);
        accz[0] = __builtin_amdgcn_wmma_f32_16x16x32_bf16(false, a0.v, false, bzf.v, (short)0, accz[0], false, false);
        accn[0] = __builtin_amdgcn_wmma_f32_16x16x32_bf16(false, a0.v, false, bnf.v, (short)0, accn[0], false, false);
        accr[1] = __builtin_amdgcn_wmma_f32_16x16x32_bf16(false, a1.v, false, brf.v, (short)0, accr[1], false, false);
        accz[1] = __builtin_amdgcn_wmma_f32_16x16x32_bf16(false, a1.v, false, bzf.v, (short)0, accz[1], false, false);
        accn[1] = __builtin_amdgcn_wmma_f32_16x16x32_bf16(false, a1.v, false, bnf.v, (short)0, accn[1], false, false);
      }

#pragma unroll
      for (int mt = 0; mt < 2; ++mt) {
        const int rowb = bBase + mt*16 + 8*half;
        F8 xr, xz, xn;
        const f32x4* pr = (const f32x4*)(XiT + (size_t)colh*B_ + rowb);
        const f32x4* pz = (const f32x4*)(XiT + (size_t)(HID_  + colh)*B_ + rowb);
        const f32x4* pn = (const f32x4*)(XiT + (size_t)(2*HID_ + colh)*B_ + rowb);
        xr.q[0] = pr[0]; xr.q[1] = pr[1];
        xz.q[0] = pz[0]; xz.q[1] = pz[1];
        xn.q[0] = pn[0]; xn.q[1] = pn[1];
#pragma unroll
        for (int r = 0; r < 8; ++r) {
          const float rg = sigm(xr.f[r] + accr[mt][r] + bhr[ntl]);
          const float zg = sigm(xz.f[r] + accz[mt][r] + bhz[ntl]);
          const float ng = tanhf(xn.f[r] + rg*(accn[mt][r] + bhn[ntl]));
          const float hn = (1.0f - zg)*ng + zg*hreg[mt][ntl][r];
          hreg[mt][ntl][r] = hn;
          hbuf[nxt][mt*16 + 8*half + r][colh] = f2bf(hn);
        }
      }
    }
    __syncthreads();   // h(t) fully in hbuf[nxt]

    // ---- phase 2 (waves 0-3): actions[:,t,:] = tanh(h @ Wo + bo) ----
    if (w < 4) {
      const int mt = w >> 1, at = w & 1;
      v8f acc = vzero;
      for (int kt = 0; kt < 16; ++kt) {
        const int kb = kt*32 + 16*half;
        FragAB a, bw;
        const u16* ap = &hbuf[nxt][mt*16 + lcol][kb];
        a.q[0] = *(const u32x4*)(ap); a.q[1] = *(const u32x4*)(ap + 8);
        const u16* bp = wo_swz + ((size_t)(kt*2 + at)*32 + lane)*16;
        bw.q[0] = *(const u32x4*)(bp); bw.q[1] = *(const u32x4*)(bp + 8);
        acc = __builtin_amdgcn_wmma_f32_16x16x32_bf16(false, a.v, false, bw.v, (short)0, acc, false, false);
      }
      const int colA = at*16 + lcol;
#pragma unroll
      for (int r = 0; r < 8; ++r) {
        const int row = bBase + mt*16 + 8*half + r;
        out[(size_t)row*(CTX_*ACT_) + t*ACT_ + colA] = tanhf(acc[r] + bo_l);
      }
    }
    cur = nxt;
  }
}

extern "C" void kernel_launch(void* const* d_in, const int* in_sizes, int n_in,
                              void* d_out, int out_size, void* d_ws, size_t ws_size,
                              hipStream_t stream) {
  (void)in_sizes; (void)n_in; (void)out_size; (void)ws_size;
  const float* s_t = (const float*)d_in[0];
  const float* z_t = (const float*)d_in[1];
  const float* W1  = (const float*)d_in[2];
  const float* b1  = (const float*)d_in[3];
  const float* g1  = (const float*)d_in[4];
  const float* be1 = (const float*)d_in[5];
  const float* W2  = (const float*)d_in[6];
  const float* b2  = (const float*)d_in[7];
  const float* g2  = (const float*)d_in[8];
  const float* be2 = (const float*)d_in[9];
  const float* W3  = (const float*)d_in[10];
  const float* b3  = (const float*)d_in[11];
  const float* Wi  = (const float*)d_in[12];
  const float* Wh  = (const float*)d_in[13];
  const float* bh  = (const float*)d_in[14];
  const float* Wo  = (const float*)d_in[15];
  const float* bo  = (const float*)d_in[16];
  float* out = (float*)d_out;
  char*  ws  = (char*)d_ws;

  u16*   wh_swz = (u16*)(ws + OFF_WHSWZ);
  u16*   wo_swz = (u16*)(ws + OFF_WOSWZ);
  float* x0  = (float*)(ws + OFF_X0);
  float* h1  = (float*)(ws + OFF_H1);
  float* h2  = (float*)(ws + OFF_H2);
  float* h0  = (float*)(ws + OFF_H0);
  float* xit = (float*)(ws + OFF_XIT);

  k_x0<<<B_, DIN_, 0, stream>>>(s_t, z_t, x0);
  k_swz_wh<<<192, 256, 0, stream>>>(Wh, wh_swz);
  k_swz_wo<<<4, 256, 0, stream>>>(Wo, wo_swz);
  k_mlp_ln<<<B_, MLPH_, 0, stream>>>(x0, DIN_, W1, b1, g1, be1, h1);
  k_mlp_ln<<<B_, MLPH_, 0, stream>>>(h1, MLPH_, W2, b2, g2, be2, h2);
  k_h0<<<B_, 256, 0, stream>>>(h2, W3, b3, h0);
  k_xi<<<TH3_, 256, 0, stream>>>(x0, Wi, xit);
  k_gru<<<B_/32, 256, 0, stream>>>(h0, xit, bh, wh_swz, wo_swz, bo, out);
}
